// MultiheadFastformerAttention_42855183680152
// MI455X (gfx1250) — matrix operand order repaired and checked by reading, not compiled, not run on hardware
//
#include <hip/hip_runtime.h>
#include <hip/hip_bf16.h>
#include <math.h>

typedef _Float16 v16h __attribute__((ext_vector_type(16)));
typedef _Float16 v8h  __attribute__((ext_vector_type(8)));
typedef float    v8f  __attribute__((ext_vector_type(8)));
typedef unsigned int v4u __attribute__((ext_vector_type(4)));
typedef int          v4i __attribute__((ext_vector_type(4)));
typedef int          v8i __attribute__((ext_vector_type(8)));

#define SEQ_B  16
#define SEQ_N  2048
#define EMB    1024
#define HEADS  16
#define HDIM   64
#define ROWS   (SEQ_B * SEQ_N)   /* 32768 */
#define LN_EPS 1e-5f

#if defined(__gfx1250__) && __has_builtin(__builtin_amdgcn_tensor_load_to_lds) && \
    __has_builtin(__builtin_amdgcn_s_wait_tensorcnt)
#define USE_TDM 1
#endif

// ---------------- wave32 / block reductions ----------------
__device__ __forceinline__ float wave_sum(float v) {
#pragma unroll
    for (int m = 16; m >= 1; m >>= 1) v += __shfl_xor(v, m, 32);
    return v;
}
__device__ __forceinline__ float wave_max(float v) {
#pragma unroll
    for (int m = 16; m >= 1; m >>= 1) v = fmaxf(v, __shfl_xor(v, m, 32));
    return v;
}
__device__ __forceinline__ float block_sum(float v, float* red) {
    const int lane = threadIdx.x & 31, wv = threadIdx.x >> 5;
    v = wave_sum(v);
    if (lane == 0) red[wv] = v;
    __syncthreads();
    float r = red[0];
#pragma unroll
    for (int i = 1; i < 8; ++i) r += red[i];
    __syncthreads();
    return r;
}
__device__ __forceinline__ float block_max(float v, float* red) {
    const int lane = threadIdx.x & 31, wv = threadIdx.x >> 5;
    v = wave_max(v);
    if (lane == 0) red[wv] = v;
    __syncthreads();
    float r = red[0];
#pragma unroll
    for (int i = 1; i < 8; ++i) r = fmaxf(r, red[i]);
    __syncthreads();
    return r;
}

// ---------------- LayerNorm -> f16 (one block per row of 1024) ----------------
__global__ __launch_bounds__(256) void ln_to_f16(const float* __restrict__ x,
                                                 const float* __restrict__ g,
                                                 const float* __restrict__ b,
                                                 _Float16* __restrict__ out) {
    __shared__ float red[8];
    const int row = blockIdx.x;
    const int t = threadIdx.x;
    const float4 v = ((const float4*)(x + (size_t)row * EMB))[t];

    float s = v.x + v.y + v.z + v.w;
    s = block_sum(s, red);
    const float mu = s * (1.0f / EMB);

    const float dx = v.x - mu, dy = v.y - mu, dz = v.z - mu, dw = v.w - mu;
    float sq = dx * dx + dy * dy + dz * dz + dw * dw;
    sq = block_sum(sq, red);
    const float rstd = rsqrtf(sq * (1.0f / EMB) + LN_EPS);

    const float4 gw = ((const float4*)g)[t];
    const float4 bb = ((const float4*)b)[t];
    _Float16* o = out + (size_t)row * EMB + t * 4;
    o[0] = (_Float16)(dx * rstd * gw.x + bb.x);
    o[1] = (_Float16)(dy * rstd * gw.y + bb.y);
    o[2] = (_Float16)(dz * rstd * gw.z + bb.z);
    o[3] = (_Float16)(dw * rstd * gw.w + bb.w);
}

// ---------------- weight transpose: W[n,k] f32 -> WT[k,n] f16 ----------------
__global__ __launch_bounds__(256) void transpose_w_f16(const float* __restrict__ W,
                                                       _Float16* __restrict__ WT) {
    __shared__ _Float16 tile[32][33];
    const int tx = threadIdx.x & 31;
    const int ty = threadIdx.x >> 5;   // 0..7
    const int c0 = blockIdx.x * 32;    // k (column of W)
    const int r0 = blockIdx.y * 32;    // n (row of W)
#pragma unroll
    for (int i = 0; i < 4; ++i) {
        const int r = ty + i * 8;
        tile[r][tx] = (_Float16)W[(size_t)(r0 + r) * EMB + c0 + tx];
    }
    __syncthreads();
#pragma unroll
    for (int i = 0; i < 4; ++i) {
        const int r = ty + i * 8;
        WT[(size_t)(c0 + r) * EMB + r0 + tx] = tile[tx][r];
    }
}

#if defined(USE_TDM)
// ------------- Tensor Data Mover: 2D tile (global -> LDS), D# per CDNA5 ISA --
// group1 word0 = workgroup_mask(0) | data_size<<16 | pad_enable<<20
//              | pad_interval<<22 | pad_amount<<25
// A tiles: rows of 32 f16 (64B = 16 dwords): interval 2^4 dwords (enc 3), pad 4 dwords (enc 3)
#define TDM_G1W0_A 0x06D10000u
// B tiles: rows of 128 f16 (256B = 64 dwords): interval 2^6 dwords (enc 5), pad 4 dwords (enc 3)
#define TDM_G1W0_B 0x07510000u

__device__ __forceinline__ void tdm_load_2d(unsigned lds_off, const void* gptr,
                                            unsigned tile_w, unsigned tile_h,
                                            unsigned row_stride_elems, unsigned g1w0) {
    const unsigned long long ga = (unsigned long long)gptr;
    const unsigned d0 = 1u << 24, d1 = 1u << 24;  // tensor dims (OOB bound only)
    v4u g0;
    g0[0] = 1u;                                               // count=1, user mode
    g0[1] = lds_off;                                          // LDS byte address
    g0[2] = (unsigned)ga;                                     // global_addr[31:0]
    g0[3] = (unsigned)((ga >> 32) & 0x1FFFFFFu) | 0x80000000u; // addr[56:32] | type=2
    v8i g1;
    g1[0] = (int)g1w0;
    g1[1] = (int)((d0 & 0xFFFFu) << 16);                      // dim0[15:0]
    g1[2] = (int)((d0 >> 16) | ((d1 & 0xFFFFu) << 16));       // dim0[31:16] | dim1[15:0]
    g1[3] = (int)((d1 >> 16) | (tile_w << 16));               // dim1[31:16] | tile_dim0
    g1[4] = (int)tile_h;                                      // tile_dim1 (tile_dim2=0)
    g1[5] = (int)row_stride_elems;                            // dim0_stride[31:0]
    g1[6] = 0;                                                // stride hi / dim1_stride lo
    g1[7] = 0;
    const v4i z4 = (v4i){0, 0, 0, 0};
#if __clang_major__ >= 23
    const v8i z8 = (v8i){0, 0, 0, 0, 0, 0, 0, 0};
    __builtin_amdgcn_tensor_load_to_lds(g0, g1, z4, z4, z8, 0);
#else
    __builtin_amdgcn_tensor_load_to_lds(g0, g1, z4, z4, 0);
#endif
}
#endif  // USE_TDM

// ---------------- WMMA GEMM: C[M,N] = A[M,K](f16) * WT[K,N](f16), f32 accum --
// Block: 256 thr = 8 waves, wave tile 32x64, block tile 128x128, K step 32.
// TDM path: double-buffered LDS tiles DMA'd by wave 0, TENSORcnt-synchronized.
// EPI: 0 = store f32, 1 = store f16, 2 = store f32 + bias[n] + resid[m,n] (NT)
template <int EPI>
__global__ __launch_bounds__(256) void gemm_wmma(const _Float16* __restrict__ A,
                                                 const _Float16* __restrict__ BT,
                                                 void* __restrict__ Cv,
                                                 const float* __restrict__ bias,
                                                 const float* __restrict__ resid,
                                                 int M, int Ncols, int K) {
    (void)M;
    const int lane = threadIdx.x & 31;
    const int wave = threadIdx.x >> 5;    // 0..7
    const int wm = wave & 3;              // 4 waves along M
    const int wn = wave >> 2;             // 2 waves along N
    const int bm0 = blockIdx.x * 128;
    const int bn0 = blockIdx.y * 128;
    const int bm = bm0 + wm * 32;
    const int bn = bn0 + wn * 64;

    v8f acc[2][4];
#pragma unroll
    for (int i = 0; i < 2; ++i)
#pragma unroll
        for (int j = 0; j < 4; ++j) acc[i][j] = (v8f){};

    // CDNA5 16-bit A 16x32 layout: lane<16 -> K 0..7 & 16..23 ; lane>=16 -> K 8..15 & 24..31
    const int arow  = lane & 15;
    const int akoff = (lane >> 4) << 3;

#if defined(USE_TDM)
    // LDS row padding from TDM pad feature: A 32+8 halves, B 128+8 halves
    constexpr int ASTRIDE = 40;
    constexpr int BSTRIDE = 136;
    __shared__ _Float16 ldsA[2][128 * ASTRIDE];
    __shared__ _Float16 ldsB[2][32 * BSTRIDE];

    const int steps = K / 32;
    if (threadIdx.x < 32) {  // wave 0 drives the TDM (EXEC ignored by TDM)
        tdm_load_2d((unsigned)(size_t)&ldsA[0][0], A + (size_t)bm0 * K, 32, 128,
                    (unsigned)K, TDM_G1W0_A);
        tdm_load_2d((unsigned)(size_t)&ldsB[0][0], BT + bn0, 128, 32,
                    (unsigned)Ncols, TDM_G1W0_B);
    }
    for (int s = 0; s < steps; ++s) {
        const int buf = s & 1;
        if (threadIdx.x < 32) {
            if (s + 1 < steps) {
                const int k1 = (s + 1) * 32;
                tdm_load_2d((unsigned)(size_t)&ldsA[buf ^ 1][0],
                            A + (size_t)bm0 * K + k1, 32, 128, (unsigned)K, TDM_G1W0_A);
                tdm_load_2d((unsigned)(size_t)&ldsB[buf ^ 1][0],
                            BT + (size_t)k1 * Ncols + bn0, 128, 32, (unsigned)Ncols,
                            TDM_G1W0_B);
                __builtin_amdgcn_s_wait_tensorcnt(2);  // current buffer's pair done
            } else {
                __builtin_amdgcn_s_wait_tensorcnt(0);
            }
        }
        __syncthreads();   // tiles for step s visible to all waves

        v16h af[2];
#pragma unroll
        for (int i = 0; i < 2; ++i) {
            const _Float16* ap = &ldsA[buf][(wm * 32 + i * 16 + arow) * ASTRIDE + akoff];
            v8h lo = *(const v8h*)ap;
            v8h hi = *(const v8h*)(ap + 16);
            af[i] = __builtin_shufflevector(lo, hi, 0, 1, 2, 3, 4, 5, 6, 7,
                                            8, 9, 10, 11, 12, 13, 14, 15);
        }
        v16h bfr[4];
#pragma unroll
        for (int j = 0; j < 4; ++j) {
            const _Float16* bp = &ldsB[buf][akoff * BSTRIDE + wn * 64 + j * 16 + arow];
#pragma unroll
            for (int s = 0; s < 8; ++s) {
                bfr[j][s] = bp[s * BSTRIDE];
                bfr[j][s + 8] = bp[(s + 16) * BSTRIDE];
            }
        }
#pragma unroll
        for (int i = 0; i < 2; ++i)
#pragma unroll
            for (int j = 0; j < 4; ++j)
                acc[i][j] = __builtin_amdgcn_wmma_f32_16x16x32_f16(
                    false, af[i], false, bfr[j], (short)0, acc[i][j], false, false);

        __syncthreads();   // all waves done with buf before it is re-DMA'd
    }
#else
    for (int k0 = 0; k0 < K; k0 += 32) {
        v16h af[2];
#pragma unroll
        for (int i = 0; i < 2; ++i) {
            const _Float16* ap = A + (size_t)(bm + i * 16 + arow) * K + k0 + akoff;
            v8h lo = *(const v8h*)ap;
            v8h hi = *(const v8h*)(ap + 16);
            af[i] = __builtin_shufflevector(lo, hi, 0, 1, 2, 3, 4, 5, 6, 7,
                                            8, 9, 10, 11, 12, 13, 14, 15);
        }
        const _Float16* bp0 = BT + (size_t)(k0 + akoff) * Ncols + bn + arow;
        if (k0 + 32 < K)
            __builtin_prefetch(bp0 + (size_t)32 * Ncols, 0, 3);
        v16h bfr[4];
#pragma unroll
        for (int j = 0; j < 4; ++j)
#pragma unroll
            for (int s = 0; s < 8; ++s) {
                bfr[j][s] = bp0[(size_t)s * Ncols + j * 16];
                bfr[j][s + 8] = bp0[(size_t)(s + 16) * Ncols + j * 16];
            }
#pragma unroll
        for (int i = 0; i < 2; ++i)
#pragma unroll
            for (int j = 0; j < 4; ++j)
                acc[i][j] = __builtin_amdgcn_wmma_f32_16x16x32_f16(
                    false, af[i], false, bfr[j], (short)0, acc[i][j], false, false);
    }
#endif

    // C/D layout: VGPR r: lanes 0-15 -> M=r, N=lane; lanes 16-31 -> M=r+8, N=lane-16
    const int crow = (lane >> 4) << 3;
    const int ccol = lane & 15;
#pragma unroll
    for (int i = 0; i < 2; ++i)
#pragma unroll
        for (int j = 0; j < 4; ++j) {
            const size_t row0 = (size_t)bm + i * 16 + crow;
            const int col = bn + j * 16 + ccol;
#pragma unroll
            for (int r = 0; r < 8; ++r) {
                const size_t idx = (row0 + r) * (size_t)Ncols + col;
                const float val = acc[i][j][r];
                if (EPI == 0) {
                    ((float*)Cv)[idx] = val;
                } else if (EPI == 1) {
                    ((_Float16*)Cv)[idx] = (_Float16)val;
                } else {
                    __builtin_nontemporal_store(val + bias[col] + resid[idx],
                                                &((float*)Cv)[idx]);
                }
            }
        }
}

// ---------------- global-query pooling: one block per (b,h) -----------------
__global__ __launch_bounds__(256) void pool_global_q(const float* __restrict__ q,
                                                     const float* __restrict__ wq,
                                                     float* __restrict__ gq) {
    __shared__ float sc[SEQ_N];
    __shared__ float red[8];
    __shared__ float wsh[HDIM];
    __shared__ float part[256];
    const int b = blockIdx.x >> 4, h = blockIdx.x & 15;
    const int t = threadIdx.x;
    const float* qb = q + ((size_t)b * SEQ_N) * EMB + h * HDIM;

    if (t < HDIM) wsh[t] = wq[h * HDIM + t];
    __syncthreads();

    for (int n = t; n < SEQ_N; n += 256) {
        const float* qr = qb + (size_t)n * EMB;
        float s = 0.f;
#pragma unroll 16
        for (int d = 0; d < HDIM; ++d) s += qr[d] * wsh[d];
        sc[n] = s;
    }
    __syncthreads();

    float m = -1e30f;
    for (int n = t; n < SEQ_N; n += 256) m = fmaxf(m, sc[n]);
    m = block_max(m, red);

    float sum = 0.f;
    for (int n = t; n < SEQ_N; n += 256) {
        const float e = __expf(sc[n] - m);
        sc[n] = e;
        sum += e;
    }
    sum = block_sum(sum, red);
    const float inv = 1.f / sum;

    const int d = t & 63, quarter = t >> 6;
    float acc = 0.f;
    for (int n = quarter * (SEQ_N / 4); n < (quarter + 1) * (SEQ_N / 4); ++n)
        acc += sc[n] * qb[(size_t)n * EMB + d];
    part[t] = acc;
    __syncthreads();
    if (t < HDIM)
        gq[blockIdx.x * HDIM + t] =
            (part[t] + part[t + 64] + part[t + 128] + part[t + 192]) * inv;
}

// ---------------- global-key pooling (k stored f16) -------------------------
__global__ __launch_bounds__(256) void pool_global_k(const _Float16* __restrict__ k,
                                                     const float* __restrict__ wk,
                                                     const float* __restrict__ gq,
                                                     float* __restrict__ gk) {
    __shared__ float sc[SEQ_N];
    __shared__ float red[8];
    __shared__ float wsh[HDIM];   // wk[h,:] * gq[b,h,:] fused
    __shared__ float gqs[HDIM];
    __shared__ float part[256];
    const int b = blockIdx.x >> 4, h = blockIdx.x & 15;
    const int t = threadIdx.x;
    const _Float16* kb = k + ((size_t)b * SEQ_N) * EMB + h * HDIM;

    if (t < HDIM) {
        const float g = gq[blockIdx.x * HDIM + t];
        gqs[t] = g;
        wsh[t] = g * wk[h * HDIM + t];
    }
    __syncthreads();

    for (int n = t; n < SEQ_N; n += 256) {
        const _Float16* kr = kb + (size_t)n * EMB;
        float s = 0.f;
#pragma unroll 16
        for (int d = 0; d < HDIM; ++d) s += (float)kr[d] * wsh[d];
        sc[n] = s;
    }
    __syncthreads();

    float m = -1e30f;
    for (int n = t; n < SEQ_N; n += 256) m = fmaxf(m, sc[n]);
    m = block_max(m, red);

    float sum = 0.f;
    for (int n = t; n < SEQ_N; n += 256) {
        const float e = __expf(sc[n] - m);
        sc[n] = e;
        sum += e;
    }
    sum = block_sum(sum, red);
    const float inv = 1.f / sum;

    const int d = t & 63, quarter = t >> 6;
    float acc = 0.f;
    for (int n = quarter * (SEQ_N / 4); n < (quarter + 1) * (SEQ_N / 4); ++n)
        acc += sc[n] * (float)kb[(size_t)n * EMB + d];
    part[t] = acc;
    __syncthreads();
    if (t < HDIM)   // gk[d] = gq[d] * sum_n beta_n k[n,d]
        gk[blockIdx.x * HDIM + t] =
            gqs[t] * (part[t] + part[t + 64] + part[t + 128] + part[t + 192]) * inv;
}

// ---------------- u = (gk[:,:,None,:] * v).reshape(B,N,E), raw reshape ------
__global__ __launch_bounds__(256) void build_u(const _Float16* __restrict__ v,
                                               const float* __restrict__ gk,
                                               _Float16* __restrict__ u) {
    const size_t i = (size_t)blockIdx.x * 256 + threadIdx.x;  // over B*N*E
    const int b = (int)(i >> 21);
    const int rem = (int)(i & ((1u << 21) - 1));
    const int mp = rem >> 10;          // m' in [0,2048)
    const int e = rem & 1023;          // e' in [0,1024)
    const int h = mp >> 7;
    const int n = ((mp & 127) << 4) | (e >> 6);
    const int d = e & 63;
    const float val = gk[(b * HEADS + h) * HDIM + d] *
                      (float)v[((size_t)(b * SEQ_N + n)) * EMB + h * HDIM + d];
    u[i] = (_Float16)val;
}

// ---------------- orchestration ---------------------------------------------
extern "C" void kernel_launch(void* const* d_in, const int* in_sizes, int n_in,
                              void* d_out, int out_size, void* d_ws, size_t ws_size,
                              hipStream_t stream) {
    (void)in_sizes; (void)n_in; (void)out_size; (void)ws_size;
    const float* x   = (const float*)d_in[0];
    const float* Wq  = (const float*)d_in[1];
    const float* Wk  = (const float*)d_in[2];
    const float* Wv  = (const float*)d_in[3];
    const float* Wf  = (const float*)d_in[4];
    const float* bfv = (const float*)d_in[5];
    const float* wq  = (const float*)d_in[6];
    const float* wk  = (const float*)d_in[7];
    const float* lnw = (const float*)d_in[8];
    const float* lnb = (const float*)d_in[9];

    char* ws = (char*)d_ws;
    size_t off = 0;
    _Float16* xn  = (_Float16*)(ws + off); off += (size_t)ROWS * EMB * 2;
    _Float16* WqT = (_Float16*)(ws + off); off += (size_t)EMB * EMB * 2;
    _Float16* WkT = (_Float16*)(ws + off); off += (size_t)EMB * EMB * 2;
    _Float16* WvT = (_Float16*)(ws + off); off += (size_t)EMB * EMB * 2;
    _Float16* WfT = (_Float16*)(ws + off); off += (size_t)EMB * EMB * 2;
    float*    qb  = (float*)   (ws + off); off += (size_t)ROWS * EMB * 4;
    _Float16* kb  = (_Float16*)(ws + off); off += (size_t)ROWS * EMB * 2;
    _Float16* vb  = (_Float16*)(ws + off); off += (size_t)ROWS * EMB * 2;
    _Float16* ub  = (_Float16*)(ws + off); off += (size_t)ROWS * EMB * 2;
    float*    gq  = (float*)   (ws + off); off += (size_t)SEQ_B * HEADS * HDIM * 4;
    float*    gk  = (float*)   (ws + off); off += (size_t)SEQ_B * HEADS * HDIM * 4;

    ln_to_f16<<<ROWS, 256, 0, stream>>>(x, lnw, lnb, xn);

    dim3 tg(EMB / 32, EMB / 32);
    transpose_w_f16<<<tg, 256, 0, stream>>>(Wq, WqT);
    transpose_w_f16<<<tg, 256, 0, stream>>>(Wk, WkT);
    transpose_w_f16<<<tg, 256, 0, stream>>>(Wv, WvT);
    transpose_w_f16<<<tg, 256, 0, stream>>>(Wf, WfT);

    dim3 gg(ROWS / 128, EMB / 128);
    gemm_wmma<0><<<gg, 256, 0, stream>>>(xn, WqT, qb, nullptr, nullptr, ROWS, EMB, EMB);
    gemm_wmma<1><<<gg, 256, 0, stream>>>(xn, WkT, kb, nullptr, nullptr, ROWS, EMB, EMB);
    gemm_wmma<1><<<gg, 256, 0, stream>>>(xn, WvT, vb, nullptr, nullptr, ROWS, EMB, EMB);

    pool_global_q<<<SEQ_B * HEADS, 256, 0, stream>>>(qb, wq, gq);
    pool_global_k<<<SEQ_B * HEADS, 256, 0, stream>>>(kb, wk, gq, gk);

    build_u<<<(ROWS * (size_t)EMB) / 256, 256, 0, stream>>>(vb, gk, ub);

    gemm_wmma<2><<<gg, 256, 0, stream>>>(ub, WfT, d_out, bfv, qb, ROWS, EMB, EMB);
}
